// GumbelSoftmaxVQVAE_86449101734132
// MI455X (gfx1250) — compile-verified, matrix operand order checked by reading, not run on hardware
//
#include <hip/hip_runtime.h>

typedef __attribute__((ext_vector_type(16))) _Float16 v16h;
typedef __attribute__((ext_vector_type(8)))  _Float16 v8h;
typedef __attribute__((ext_vector_type(8)))  float    v8f;

#define BB   16384
#define DIN  8192
#define HDEC 256
#define NEXP 8

static __device__ __forceinline__ v16h cat16(v8h a, v8h b) {
  return __builtin_shufflevector(a, b, 0,1,2,3,4,5,6,7,8,9,10,11,12,13,14,15);
}

// mish(x) = x * tanh(softplus(x)) = x * ((1+e^x)^2 - 1) / ((1+e^x)^2 + 1)
// 1 transcendental + ~5 VALU ops instead of a tanhf/log1pf library chain.
static __device__ __forceinline__ float mishf(float x) {
  float p  = 1.f + __expf(x);
  float p2 = p * p;
  float r  = x * (p2 - 1.f) * __builtin_amdgcn_rcpf(p2 + 1.f);
  return (x > 20.f) ? x : r;   // avoid inf/inf for large x (mish(x)->x)
}

static __device__ __forceinline__ v8f wmma16(v16h a, v16h b, v8f c) {
  // D = A(16x32 f16) * B(32x16 f16) + C(16x16 f32)
  return __builtin_amdgcn_wmma_f32_16x16x32_f16(false, a, false, b, (short)0, c, false, false);
}

// ---------------------------------------------------------------------------
// K0: transpose + fp32->f16 convert all GEMM weights into workspace.
//  wcatT [48][8192]   : cols 0..7 = enc_gate_w, 8..39 = enc_w1 (e,l), 40..47 = 0
//  w2T/w3T [8][256][256] : dec_w2/dec_w3 transposed to [e][N][K]
//  w4T [8][16][256]      : dec_w4 transposed, N padded 8->16 with zeros
// ---------------------------------------------------------------------------
__global__ void __launch_bounds__(256) k_prep(const float* __restrict__ egw,
                                              const float* __restrict__ ew1,
                                              const float* __restrict__ dw2,
                                              const float* __restrict__ dw3,
                                              const float* __restrict__ dw4,
                                              _Float16* __restrict__ wcatT,
                                              _Float16* __restrict__ w2T,
                                              _Float16* __restrict__ w3T,
                                              _Float16* __restrict__ w4T) {
  const int N0 = 48 * DIN;        // 393216
  const int N1 = 8 * HDEC * HDEC; // 524288
  const int N3 = 8 * 16 * HDEC;   // 32768
  int i = blockIdx.x * 256 + threadIdx.x;
  if (i < N0) {
    int n = i >> 13, k = i & (DIN - 1);
    float v = 0.f;
    if (n < 8)       v = egw[(size_t)k * 8 + n];
    else if (n < 40) { int e = (n - 8) >> 2, l = (n - 8) & 3;
                       v = ew1[((size_t)e * DIN + k) * 4 + l]; }
    wcatT[i] = (_Float16)v;
  } else if (i < N0 + N1) {
    int j = i - N0;
    int e = j >> 16, n = (j >> 8) & 255, k = j & 255;
    w2T[j] = (_Float16)dw2[(size_t)e * 65536 + k * 256 + n];
  } else if (i < N0 + 2 * N1) {
    int j = i - N0 - N1;
    int e = j >> 16, n = (j >> 8) & 255, k = j & 255;
    w3T[j] = (_Float16)dw3[(size_t)e * 65536 + k * 256 + n];
  } else if (i < N0 + 2 * N1 + N3) {
    int j = i - N0 - 2 * N1;
    int e = j >> 12, n = (j >> 8) & 15, k = j & 255;
    w4T[j] = (_Float16)((n < 8) ? dw4[(size_t)e * 2048 + k * 8 + n] : 0.f);
  }
}

// ---------------------------------------------------------------------------
// K1: skinny GEMM  Y[16384,48] = x[16384,8192] @ WcatT^T   (memory bound on x)
// 8 waves / block, each wave owns a 16-row M tile, 3 N-tiles of 16, K loop 32.
// ---------------------------------------------------------------------------
__global__ void __launch_bounds__(256) k_enc_gemm(const float* __restrict__ x,
                                                  const _Float16* __restrict__ wT,
                                                  float* __restrict__ Y) {
  const int lane = threadIdx.x & 31;
  const int wv   = threadIdx.x >> 5;
  const int mrow = lane & 15;
  const int hh   = lane >> 4;
  const int m0   = (blockIdx.x * 8 + wv) * 16;
  const float* xr = x + (size_t)(m0 + mrow) * DIN;
  v8f acc[3] = {};
  for (int kk = 0; kk < DIN; kk += 32) {
    const float4* p0 = (const float4*)(xr + kk + hh * 8);
    const float4* p1 = (const float4*)(xr + kk + 16 + hh * 8);
    float4 a0 = p0[0], a1 = p0[1];
    float4 b0 = p1[0], b1 = p1[1];
    v16h af;
    af[0]  = (_Float16)a0.x; af[1]  = (_Float16)a0.y; af[2]  = (_Float16)a0.z; af[3]  = (_Float16)a0.w;
    af[4]  = (_Float16)a1.x; af[5]  = (_Float16)a1.y; af[6]  = (_Float16)a1.z; af[7]  = (_Float16)a1.w;
    af[8]  = (_Float16)b0.x; af[9]  = (_Float16)b0.y; af[10] = (_Float16)b0.z; af[11] = (_Float16)b0.w;
    af[12] = (_Float16)b1.x; af[13] = (_Float16)b1.y; af[14] = (_Float16)b1.z; af[15] = (_Float16)b1.w;
    #pragma unroll
    for (int nt = 0; nt < 3; ++nt) {
      const _Float16* br = wT + (size_t)(nt * 16 + mrow) * DIN + kk;
      v16h bf = cat16(*(const v8h*)(br + hh * 8), *(const v8h*)(br + 16 + hh * 8));
      acc[nt] = wmma16(af, bf, acc[nt]);
    }
  }
  #pragma unroll
  for (int nt = 0; nt < 3; ++nt)
    #pragma unroll
    for (int r = 0; r < 8; ++r)
      Y[(size_t)(m0 + r + 8 * hh) * 48 + nt * 16 + mrow] = acc[nt][r];
}

// ---------------------------------------------------------------------------
// K2: per-row scalar epilogue: enc gate softmax, 8x tiny 4-wide expert MLPs,
// estimate, codebook logits, Gumbel softmax + straight-through one-hot,
// decoder gate softmax. Emits c[B,4], gate_d[B,8].
// ---------------------------------------------------------------------------
static __device__ __forceinline__ void ln4(float* v) {
  float m = 0.25f * (v[0] + v[1] + v[2] + v[3]);
  float var = 0.f;
  #pragma unroll
  for (int i = 0; i < 4; ++i) { float d = v[i] - m; var += d * d; }
  var *= 0.25f;
  float r = rsqrtf(var + 1e-5f);
  #pragma unroll
  for (int i = 0; i < 4; ++i) v[i] = (v[i] - m) * r;
}

__global__ void __launch_bounds__(256) k_epilogue(
    const float* __restrict__ Y,   const float* __restrict__ gum,
    const float* __restrict__ cb,  const float* __restrict__ egb,
    const float* __restrict__ eb1, const float* __restrict__ ew2,
    const float* __restrict__ eb2, const float* __restrict__ ew3,
    const float* __restrict__ eb3, const float* __restrict__ ew4,
    const float* __restrict__ eb4, const float* __restrict__ dgw,
    const float* __restrict__ dgb, float* __restrict__ c_ws,
    float* __restrict__ g_ws) {
  int b = blockIdx.x * 256 + threadIdx.x;
  if (b >= BB) return;
  const float* y = Y + (size_t)b * 48;
  float gl[8]; float mx = -3.4e38f;
  #pragma unroll
  for (int j = 0; j < 8; ++j) { gl[j] = y[j] + egb[j]; mx = fmaxf(mx, gl[j]); }
  float se = 0.f;
  #pragma unroll
  for (int j = 0; j < 8; ++j) { gl[j] = __expf(gl[j] - mx); se += gl[j]; }
  float inv = 1.f / se;
  float est[4] = {0.f, 0.f, 0.f, 0.f};
  for (int e = 0; e < 8; ++e) {
    float h[4], t[4];
    #pragma unroll
    for (int l = 0; l < 4; ++l) h[l] = mishf(y[8 + e * 4 + l] + eb1[e * 4 + l]);
    ln4(h);
    #pragma unroll
    for (int k = 0; k < 4; ++k) {
      float a = eb2[e * 4 + k];
      #pragma unroll
      for (int l = 0; l < 4; ++l) a += h[l] * ew2[e * 16 + l * 4 + k];
      t[k] = mishf(a);
    }
    ln4(t);
    #pragma unroll
    for (int k = 0; k < 4; ++k) {
      float a = eb3[e * 4 + k];
      #pragma unroll
      for (int l = 0; l < 4; ++l) a += t[l] * ew3[e * 16 + l * 4 + k];
      h[k] = mishf(a);
    }
    float ge = gl[e] * inv;
    #pragma unroll
    for (int k = 0; k < 4; ++k) {
      float a = eb4[e * 4 + k];
      #pragma unroll
      for (int l = 0; l < 4; ++l) a += h[l] * ew4[e * 16 + l * 4 + k];
      est[k] += ge * a;
    }
  }
  float yl[4]; float mx2 = -3.4e38f;
  #pragma unroll
  for (int n = 0; n < 4; ++n) {
    float lg = 0.f;
    #pragma unroll
    for (int l = 0; l < 4; ++l) lg += est[l] * cb[n * 4 + l];
    float u = gum[(size_t)b * 4 + n];
    u = fminf(fmaxf(u, 1e-10f), 1.0f - 1e-7f);
    yl[n] = lg - __logf(-__logf(u));      // + gumbel, TAU = 1
    mx2 = fmaxf(mx2, yl[n]);
  }
  float ys[4]; float s2 = 0.f;
  #pragma unroll
  for (int n = 0; n < 4; ++n) { ys[n] = __expf(yl[n] - mx2); s2 += ys[n]; }
  float i2 = 1.f / s2;
  int kb = 0; float bm = -1.f;
  #pragma unroll
  for (int n = 0; n < 4; ++n) { ys[n] *= i2; if (ys[n] > bm) { bm = ys[n]; kb = n; } }
  float c[4];
  #pragma unroll
  for (int n = 0; n < 4; ++n) {
    float hard = (n == kb) ? 1.f : 0.f;
    c[n] = (hard - ys[n]) + ys[n];        // straight-through fp order
    c_ws[(size_t)b * 4 + n] = c[n];
  }
  float dl[8]; float mx3 = -3.4e38f;
  #pragma unroll
  for (int j = 0; j < 8; ++j) {
    float a = dgb[j];
    #pragma unroll
    for (int n = 0; n < 4; ++n) a += c[n] * dgw[n * 8 + j];
    dl[j] = a; mx3 = fmaxf(mx3, a);
  }
  float s3 = 0.f;
  #pragma unroll
  for (int j = 0; j < 8; ++j) { dl[j] = __expf(dl[j] - mx3); s3 += dl[j]; }
  float i3 = 1.f / s3;
  #pragma unroll
  for (int j = 0; j < 8; ++j) g_ws[(size_t)b * 8 + j] = dl[j] * i3;
}

// ---------------------------------------------------------------------------
// K3: fused decoder MoE. 4 waves/block; each wave owns 16 rows; loops 8 experts
// with h kept in LDS ping-pong buffers (f16). All GEMMs are WMMA 16x16x32 f16.
// ---------------------------------------------------------------------------
static __device__ __forceinline__ void ln_rows(const _Float16* __restrict__ In,
                                               _Float16* __restrict__ Outp, int lane) {
  for (int r = 0; r < 16; ++r) {
    const v8h pv = *(const v8h*)(In + r * HDEC + lane * 8);
    float v[8]; float s = 0.f, ss = 0.f;
    #pragma unroll
    for (int i = 0; i < 8; ++i) { v[i] = (float)pv[i]; s += v[i]; ss += v[i] * v[i]; }
    #pragma unroll
    for (int m = 16; m >= 1; m >>= 1) { s += __shfl_xor(s, m, 32); ss += __shfl_xor(ss, m, 32); }
    float mean = s * (1.f / 256.f);
    float var  = ss * (1.f / 256.f) - mean * mean;
    float rstd = rsqrtf(var + 1e-5f);
    v8h ov;
    #pragma unroll
    for (int i = 0; i < 8; ++i) ov[i] = (_Float16)((v[i] - mean) * rstd);
    *(v8h*)(Outp + r * HDEC + lane * 8) = ov;
  }
}

// [16,256] @ [256,256] with mish epilogue; two N-tiles in flight -> two
// independent WMMA accumulation chains to hide WMMA latency.
static __device__ __forceinline__ void gemm_mish(const _Float16* __restrict__ A,
                                                 const _Float16* __restrict__ BT,
                                                 const float* __restrict__ bias,
                                                 _Float16* __restrict__ Out,
                                                 int mrow, int hh) {
  v16h af[8];
  #pragma unroll
  for (int kt = 0; kt < 8; ++kt) {
    const _Float16* ar = A + mrow * HDEC + kt * 32;
    af[kt] = cat16(*(const v8h*)(ar + hh * 8), *(const v8h*)(ar + 16 + hh * 8));
  }
  for (int nt = 0; nt < 16; nt += 2) {
    v8f acc0 = {}, acc1 = {};
    const _Float16* br0 = BT + (size_t)(nt * 16 + mrow) * HDEC;
    const _Float16* br1 = br0 + (size_t)16 * HDEC;
    #pragma unroll
    for (int kt = 0; kt < 8; ++kt) {
      v16h bf0 = cat16(*(const v8h*)(br0 + kt * 32 + hh * 8),
                       *(const v8h*)(br0 + kt * 32 + 16 + hh * 8));
      v16h bf1 = cat16(*(const v8h*)(br1 + kt * 32 + hh * 8),
                       *(const v8h*)(br1 + kt * 32 + 16 + hh * 8));
      acc0 = wmma16(af[kt], bf0, acc0);
      acc1 = wmma16(af[kt], bf1, acc1);
    }
    float bv0 = bias[nt * 16 + mrow];
    float bv1 = bias[nt * 16 + 16 + mrow];
    #pragma unroll
    for (int r = 0; r < 8; ++r) {
      Out[(r + 8 * hh) * HDEC + nt * 16 + mrow]      = (_Float16)mishf(acc0[r] + bv0);
      Out[(r + 8 * hh) * HDEC + nt * 16 + 16 + mrow] = (_Float16)mishf(acc1[r] + bv1);
    }
  }
}

__global__ void __launch_bounds__(128) k_dec(const float* __restrict__ c_ws,
                                             const float* __restrict__ g_ws,
                                             const float* __restrict__ dw1,
                                             const float* __restrict__ db1,
                                             const _Float16* __restrict__ w2T,
                                             const float* __restrict__ db2,
                                             const _Float16* __restrict__ w3T,
                                             const float* __restrict__ db3,
                                             const _Float16* __restrict__ w4T,
                                             const float* __restrict__ db4,
                                             float* __restrict__ out) {
  extern __shared__ char smem[];
  const int lane = threadIdx.x & 31;
  const int wv   = threadIdx.x >> 5;
  const int mrow = lane & 15;
  const int hh   = lane >> 4;
  const int m0   = (blockIdx.x * 4 + wv) * 16;
  _Float16* bufA = (_Float16*)(smem + wv * 16384);
  _Float16* bufB = (_Float16*)(smem + wv * 16384 + 8192);
  float*    ct   = (float*)(smem + 65536 + wv * 256);
  for (int i = lane; i < 64; i += 32) ct[i] = c_ws[(size_t)m0 * 4 + i];
  __syncthreads();
  v8f recon = {};
  for (int e = 0; e < NEXP; ++e) {
    // h1 = mish(one_hot @ dec_w1[e] + b1)  -> bufB (pre-LN)
    for (int idx = lane; idx < 16 * HDEC; idx += 32) {
      int m = idx >> 8, n = idx & 255;
      float v = db1[e * HDEC + n];
      #pragma unroll
      for (int k = 0; k < 4; ++k) v += ct[m * 4 + k] * dw1[(e * 4 + k) * HDEC + n];
      bufB[m * HDEC + n] = (_Float16)mishf(v);
    }
    __syncthreads();
    ln_rows(bufB, bufA, lane);
    __syncthreads();
    gemm_mish(bufA, w2T + (size_t)e * 65536, db2 + e * HDEC, bufB, mrow, hh);
    __syncthreads();
    ln_rows(bufB, bufA, lane);
    __syncthreads();
    gemm_mish(bufA, w3T + (size_t)e * 65536, db3 + e * HDEC, bufB, mrow, hh);
    __syncthreads();
    // GEMM3: [16,256] @ [256,8 (pad 16)] + b4, gate-weighted accumulate
    v16h af[8];
    #pragma unroll
    for (int kt = 0; kt < 8; ++kt) {
      const _Float16* ar = bufB + mrow * HDEC + kt * 32;
      af[kt] = cat16(*(const v8h*)(ar + hh * 8), *(const v8h*)(ar + 16 + hh * 8));
    }
    v8f acc = {};
    const _Float16* brow = w4T + (size_t)e * 4096 + (size_t)mrow * HDEC;
    #pragma unroll
    for (int kt = 0; kt < 8; ++kt) {
      v16h bf = cat16(*(const v8h*)(brow + kt * 32 + hh * 8),
                      *(const v8h*)(brow + kt * 32 + 16 + hh * 8));
      acc = wmma16(af[kt], bf, acc);
    }
    float bias = (mrow < 8) ? db4[e * 8 + mrow] : 0.f;
    #pragma unroll
    for (int r = 0; r < 8; ++r) {
      int m = m0 + r + 8 * hh;
      float g = g_ws[(size_t)m * 8 + e];
      recon[r] += g * (acc[r] + bias);
    }
    __syncthreads();
  }
  if (mrow < 8) {
    #pragma unroll
    for (int r = 0; r < 8; ++r)
      out[(size_t)(m0 + r + 8 * hh) * 8 + mrow] = recon[r];
  }
}

// ---------------------------------------------------------------------------
extern "C" void kernel_launch(void* const* d_in, const int* in_sizes, int n_in,
                              void* d_out, int out_size, void* d_ws, size_t ws_size,
                              hipStream_t stream) {
  (void)in_sizes; (void)n_in; (void)out_size; (void)ws_size;
  const float* x   = (const float*)d_in[0];
  const float* gum = (const float*)d_in[1];
  const float* cb  = (const float*)d_in[2];
  const float* egw = (const float*)d_in[3];
  const float* egb = (const float*)d_in[4];
  const float* ew1 = (const float*)d_in[5];
  const float* eb1 = (const float*)d_in[6];
  const float* ew2 = (const float*)d_in[7];
  const float* eb2 = (const float*)d_in[8];
  const float* ew3 = (const float*)d_in[9];
  const float* eb3 = (const float*)d_in[10];
  const float* ew4 = (const float*)d_in[11];
  const float* eb4 = (const float*)d_in[12];
  const float* dgw = (const float*)d_in[13];
  const float* dgb = (const float*)d_in[14];
  const float* dw1 = (const float*)d_in[15];
  const float* db1 = (const float*)d_in[16];
  const float* dw2 = (const float*)d_in[17];
  const float* db2 = (const float*)d_in[18];
  const float* dw3 = (const float*)d_in[19];
  const float* db3 = (const float*)d_in[20];
  const float* dw4 = (const float*)d_in[21];
  const float* db4 = (const float*)d_in[22];

  char* ws = (char*)d_ws;
  _Float16* wcatT = (_Float16*)(ws + 0);        //   786432 B
  _Float16* w2T   = (_Float16*)(ws + 786432);   //  1048576 B
  _Float16* w3T   = (_Float16*)(ws + 1835008);  //  1048576 B
  _Float16* w4T   = (_Float16*)(ws + 2883584);  //    65536 B
  float*    Y     = (float*)   (ws + 2949120);  //  3145728 B
  float*    c_ws  = (float*)   (ws + 6094848);  //   262144 B
  float*    g_ws  = (float*)   (ws + 6356992);  //   524288 B

  k_prep<<<5760, 256, 0, stream>>>(egw, ew1, dw2, dw3, dw4, wcatT, w2T, w3T, w4T);
  k_enc_gemm<<<128, 256, 0, stream>>>(x, wcatT, Y);
  k_epilogue<<<64, 256, 0, stream>>>(Y, gum, cb, egb, eb1, ew2, eb2, ew3, eb3,
                                     ew4, eb4, dgw, dgb, c_ws, g_ws);
  k_dec<<<256, 128, 66560, stream>>>(c_ws, g_ws, dw1, db1, w2T, db2, w3T, db3,
                                     w4T, db4, (float*)d_out);
}
